// DeepSAD_GVAE_14929306321517
// MI455X (gfx1250) — matrix operand order repaired and checked by reading, not compiled, not run on hardware
//
#include <hip/hip_runtime.h>
#include <hip/hip_bf16.h>

// ---------------------------------------------------------------------------
// VGAE forward (GENConv softmax aggregation) for gfx1250 / MI455X.
// f32 numerics for softmax/BN/exp; GEMMs via v_wmma_f32_16x16x32_bf16 with
// operands pre-packed into bf16 fragment layout (one pass, preprocessing
// fused). K-loop fully unrolled via templated NKB: straight-line b128 clause
// loads + WMMA.
// ---------------------------------------------------------------------------

#define N_NODES  50000
#define N_EDGES  800000
#define C_IN     64
#define H_DIM    128
#define K2_DIM   256            // 2*H_DIM: hidden width of every GENConv MLP
#define MSG_EPS  1e-7f
#define BN_EPS   1e-5f

typedef __bf16 v16bf __attribute__((ext_vector_type(16)));   // one WMMA fragment / lane (32B)
typedef __bf16 v8bf  __attribute__((ext_vector_type(8)));    // 16B store unit
typedef float  v8f   __attribute__((ext_vector_type(8)));

// --------------------------- utility kernels -------------------------------

__global__ void fill_zero_kernel(float* __restrict__ p, long long n) {
  long long i = (long long)blockIdx.x * blockDim.x + threadIdx.x;
  if (i < n) p[i] = 0.0f;
}

// Per-edge softmax-aggregation passes (compile-time PASS):
//  0: m = segmax(s)      via int-bit atomicMax (s > 0 since t=1, m init 0)
//  1: den += exp(s - m)
//  2: agg += msg * exp(s - m) / (den + 1e-16)
template <int PASS>
__global__ void edge_pass_kernel(const float* __restrict__ X,
                                 const long long* __restrict__ ei,   // [2, E] int64
                                 const float* __restrict__ tptr,
                                 float* __restrict__ m,
                                 float* __restrict__ den,
                                 float* __restrict__ agg,
                                 long long E) {
  const int d  = H_DIM;
  const int dq = d >> 2;
  long long i = (long long)blockIdx.x * blockDim.x + threadIdx.x;
  long long total = E * (long long)dq;
  if (i >= total) return;
  long long e  = i / dq;
  int       f4 = (int)(i % dq) << 2;
  long long src = ei[e];
  long long dst = ei[E + e];
  const float tv = *tptr;
  float4 xv = *(const float4*)(X + src * (long long)d + f4);
  float xs[4] = {xv.x, xv.y, xv.z, xv.w};
  long long ob = dst * (long long)d + f4;
#pragma unroll
  for (int c = 0; c < 4; ++c) {
    float msg = fmaxf(xs[c], 0.0f) + MSG_EPS;
    float s   = tv * msg;
    long long o = ob + c;
    if (PASS == 0) {
      atomicMax((int*)(m + o), __float_as_int(s));   // positive-float max
    } else if (PASS == 1) {
      atomicAdd(den + o, __expf(s - m[o]));
    } else {
      atomicAdd(agg + o, msg * __expf(s - m[o]) / (den[o] + 1e-16f));
    }
  }
}

// BatchNorm train statistics: coalesced column-per-thread partial sums.
__global__ void bn_partial_kernel(const float* __restrict__ Hm,
                                  float* __restrict__ sum, float* __restrict__ sumsq,
                                  int n, int k) {
  int col = threadIdx.x;                // blockDim.x == k
  float s = 0.0f, sq = 0.0f;
  for (long long r = blockIdx.x; r < n; r += gridDim.x) {
    float v = Hm[r * (long long)k + col];
    s += v; sq += v * v;
  }
  atomicAdd(&sum[col], s);
  atomicAdd(&sumsq[col], sq);
}

__global__ void bn_finalize_kernel(const float* __restrict__ sum,
                                   const float* __restrict__ sumsq,
                                   float* __restrict__ mu, float* __restrict__ rstd,
                                   float invn, int k) {
  int j = blockIdx.x * blockDim.x + threadIdx.x;
  if (j >= k) return;
  float mv = sum[j] * invn;
  float vv = sumsq[j] * invn - mv * mv;
  mu[j]   = mv;
  rstd[j] = rsqrtf(vv + BN_EPS);
}

// Z = mu + noise * exp(min(logstd, 10))
__global__ void z_kernel(const float* __restrict__ mu, const float* __restrict__ ls,
                         const float* __restrict__ noise, float* __restrict__ Z,
                         long long n) {
  long long i = (long long)blockIdx.x * blockDim.x + threadIdx.x;
  if (i < n) Z[i] = mu[i] + noise[i] * __expf(fminf(ls[i], 10.0f));
}

// ------------------------- operand packing (bf16) --------------------------
// A-fragment layout (ISA 7.12.2, 16-bit A 16x32): per lane 16 elems; VGPR v
// holds a K pair; lanes 0-15 cover K 0-7/16-23, lanes 16-31 K 8-15/24-31.
// Packed storage: frag[(tileM*nKb + kb)*32 + lane][e], 2B elems, 32B/lane.
//  MODE 0: A     MODE 1: A + A2     MODE 2: relu((A-mu)*rstd*g + b)
template <int MODE>
__global__ void pack_a_kernel(const float* __restrict__ A,
                              const float* __restrict__ A2,
                              __bf16* __restrict__ P,
                              int n, int kin,
                              const float* __restrict__ mu,
                              const float* __restrict__ rstd,
                              const float* __restrict__ g,
                              const float* __restrict__ b) {
  long long tid = (long long)blockIdx.x * blockDim.x + threadIdx.x;
  long long total = ((long long)n * kin) >> 3;     // one thread per 8 elems
  if (tid >= total) return;
  const int nKb = kin >> 5;
  int  half = (int)(tid & 1);
  long long t2 = tid >> 1;
  int  lane = (int)(t2 & 31);
  long long tileIdx = t2 >> 5;                     // tm*nKb + kb
  int tm = (int)(tileIdx / nKb);
  int kb = (int)(tileIdx % nKb);
  int l15 = lane & 15, hi = lane >> 4;
  int row = tm * 16 + l15;
  v8bf vals;
#pragma unroll
  for (int j = 0; j < 8; ++j) {
    int e = half * 8 + j;
    int v = e >> 1, p = e & 1;
    int ka = kb * 32 + ((v < 4) ? (2 * v) : (8 + 2 * v)) + hi * 8 + p;
    float x = A[(long long)row * kin + ka];
    if (MODE == 1) x += A2[(long long)row * kin + ka];
    if (MODE == 2) { x = (x - mu[ka]) * rstd[ka] * g[ka] + b[ka]; x = fmaxf(x, 0.0f); }
    vals[j] = (__bf16)x;
  }
  *(v8bf*)(P + tid * 8) = vals;
}

// B-fragment layout: row-across-lanes; lanes 0-15 hold K 0-7/16-23 of the 32-block,
// lanes 16-31 hold K 8-15/24-31. Packed: frag[(tileN*nKb + kb)*32 + lane][e].
__global__ void pack_b_kernel(const float* __restrict__ W,
                              __bf16* __restrict__ P,
                              int kin, int kout) {
  long long tid = (long long)blockIdx.x * blockDim.x + threadIdx.x;
  long long total = ((long long)kin * kout) >> 3;
  if (tid >= total) return;
  const int nKb = kin >> 5;
  int  half = (int)(tid & 1);
  long long t2 = tid >> 1;
  int  lane = (int)(t2 & 31);
  long long tileIdx = t2 >> 5;                     // tn*nKb + kb
  int tn = (int)(tileIdx / nKb);
  int kb = (int)(tileIdx % nKb);
  int col = tn * 16 + (lane & 15);
  int hi  = lane >> 4;
  v8bf vals;
#pragma unroll
  for (int j = 0; j < 8; ++j) {
    int e  = half * 8 + j;
    int kk = kb * 32 + ((e < 8) ? e : (e + 8)) + hi * 8;
    vals[j] = (__bf16)W[(long long)kk * kout + col];
  }
  *(v8bf*)(P + tid * 8) = vals;
}

// ------------------------------ WMMA GEMM ----------------------------------
// C[n,kout] = packedA @ packedB, one wave per 16x16 tile, K fully unrolled
// (NKB in {2,4,8}). Straight-line: 2*NKB 32B fragment loads + NKB WMMAs.
//  OM 0: none   1: relu   2: min(x,10)   3: +bias[col]
template <int OM, int NKB>
__global__ void gemm_packed_wmma_kernel(const v16bf* __restrict__ pA,
                                        const v16bf* __restrict__ pB,
                                        float* __restrict__ C,
                                        int n, int kout,
                                        const float* __restrict__ bias) {
  const int lane = threadIdx.x & 31;
  const int wave = threadIdx.x >> 5;
  const int tilesK = kout >> 4;
  const long long nTiles = (long long)(n >> 4) * tilesK;
  long long tile = (long long)blockIdx.x * (blockDim.x >> 5) + wave;
  if (tile >= nTiles) return;           // wave-uniform exit: EXEC all-1s below

  const int tm = (int)(tile / tilesK);
  const int tn = (int)(tile % tilesK);
  const long long aoff = (long long)tm * NKB * 32 + lane;
  const long long boff = (long long)tn * NKB * 32 + lane;

  // Load all fragments first (compiler clauses the b128 loads), then WMMA.
  v16bf af[NKB], bv[NKB];
#pragma unroll
  for (int kb = 0; kb < NKB; ++kb) {
    af[kb] = pA[aoff + kb * 32];
    bv[kb] = pB[boff + kb * 32];
  }
  v8f acc = {0.f, 0.f, 0.f, 0.f, 0.f, 0.f, 0.f, 0.f};
#pragma unroll
  for (int kb = 0; kb < NKB; ++kb) {
    acc = __builtin_amdgcn_wmma_f32_16x16x32_bf16(
        /*neg_a=*/false, af[kb], /*neg_b=*/false, bv[kb],
        /*c_mod=*/(short)0, acc, /*reuse_a=*/false, /*reuse_b=*/false);
  }

  const int hi    = lane >> 4;
  const int bcol  = tn * 16 + (lane & 15);
  const int rbase = tm * 16 + hi * 8;   // C/D layout: VGPR v -> row v (+8 hi)
#pragma unroll
  for (int v = 0; v < 8; ++v) {
    float val = acc[v];
    if (OM == 3)      val += bias[bcol];
    else if (OM == 1) val = fmaxf(val, 0.0f);
    else if (OM == 2) val = fminf(val, 10.0f);
    C[(long long)(rbase + v) * kout + bcol] = val;
  }
}

// ------------------------------- host side ---------------------------------

struct GenP { const float *b, *g, *t, *w1, *w2; };

extern "C" void kernel_launch(void* const* d_in, const int* in_sizes, int n_in,
                              void* d_out, int out_size, void* d_ws, size_t ws_size,
                              hipStream_t stream) {
  (void)in_sizes; (void)out_size; (void)ws_size;
  if (n_in < 35) return;

  // Input order: x, edge_index, noise, then params flattened pytree-style
  // (dict keys sorted): dec[0], dec[1], enc[0], enc[1], enc_logstd, enc_mu,
  // n_lin_b, n_lin_w; each GENConv dict sorted as {b, g, t, w1, w2}.
  const float*     x     = (const float*)d_in[0];
  const long long* ei    = (const long long*)d_in[1];   // int64 [2, E]
  const float*     noise = (const float*)d_in[2];
  auto gp = [&](int base) {
    GenP p;
    p.b  = (const float*)d_in[base + 0];
    p.g  = (const float*)d_in[base + 1];
    p.t  = (const float*)d_in[base + 2];
    p.w1 = (const float*)d_in[base + 3];
    p.w2 = (const float*)d_in[base + 4];
    return p;
  };
  GenP dec0 = gp(3), dec1 = gp(8), enc0 = gp(13), enc1 = gp(18);
  GenP encls = gp(23), encmu = gp(28);
  const float* n_lin_b = (const float*)d_in[33];
  const float* n_lin_w = (const float*)d_in[34];

  const long long N = N_NODES, E = N_EDGES;

  // Workspace: h | m | den | agg | hmid | mu_out | logstd_out | stats | pA | pW
  float* ws    = (float*)d_ws;
  float* hbuf  = ws;
  float* mbuf  = hbuf + N * H_DIM;
  float* den   = mbuf + N * H_DIM;
  float* agg   = den  + N * H_DIM;
  float* hmid  = agg  + N * H_DIM;          // N * K2
  float* muo   = hmid + N * K2_DIM;
  float* lso   = muo  + N * H_DIM;
  float* stats = lso  + N * H_DIM;          // sum | sumsq | mu | rstd (K2 each)
  __bf16* pAbuf = (__bf16*)(stats + 4 * K2_DIM);            // N*K2 bf16 (32B-aligned)
  __bf16* pWbuf = pAbuf + N * K2_DIM;                       // K2*K2 bf16 max

  float* out_h = (float*)d_out;             // [N, 64]
  float* out_z = out_h + N * C_IN;          // [N, 128]

  auto nblk = [](long long cnt) { return (int)((cnt + 255) / 256); };
  auto fill0 = [&](float* p, long long cnt) {
    fill_zero_kernel<<<nblk(cnt), 256, 0, stream>>>(p, cnt);
  };
  auto pack_a = [&](int mode, const float* A, const float* A2, int n, int kin,
                    const float* mu, const float* rstd, const float* g,
                    const float* b) {
    int blocks = nblk(((long long)n * kin) >> 3);
    if (mode == 0)
      pack_a_kernel<0><<<blocks, 256, 0, stream>>>(A, A2, pAbuf, n, kin, mu, rstd, g, b);
    else if (mode == 1)
      pack_a_kernel<1><<<blocks, 256, 0, stream>>>(A, A2, pAbuf, n, kin, mu, rstd, g, b);
    else
      pack_a_kernel<2><<<blocks, 256, 0, stream>>>(A, A2, pAbuf, n, kin, mu, rstd, g, b);
  };
  auto pack_b = [&](const float* W, int kin, int kout) {
    int blocks = nblk(((long long)kin * kout) >> 3);
    pack_b_kernel<<<blocks, 256, 0, stream>>>(W, pWbuf, kin, kout);
  };
  auto gemm = [&](int om, float* C, int n, int kin, int kout, const float* bias) {
    long long tiles = (long long)(n >> 4) * (kout >> 4);
    int blocks = (int)((tiles + 7) / 8);    // 8 waves per 256-thread block
    const v16bf* pA = (const v16bf*)pAbuf;
    const v16bf* pB = (const v16bf*)pWbuf;
    int nKb = kin >> 5;                     // 2, 4 or 8
    #define LAUNCH_GEMM(OMV, NKBV) \
      gemm_packed_wmma_kernel<OMV, NKBV><<<blocks, 256, 0, stream>>>(pA, pB, C, n, kout, bias)
    if (nKb == 2) {
      switch (om) { case 0: LAUNCH_GEMM(0, 2); break; case 1: LAUNCH_GEMM(1, 2); break;
                    case 2: LAUNCH_GEMM(2, 2); break; default: LAUNCH_GEMM(3, 2); break; }
    } else if (nKb == 4) {
      switch (om) { case 0: LAUNCH_GEMM(0, 4); break; case 1: LAUNCH_GEMM(1, 4); break;
                    case 2: LAUNCH_GEMM(2, 4); break; default: LAUNCH_GEMM(3, 4); break; }
    } else {
      switch (om) { case 0: LAUNCH_GEMM(0, 8); break; case 1: LAUNCH_GEMM(1, 8); break;
                    case 2: LAUNCH_GEMM(2, 8); break; default: LAUNCH_GEMM(3, 8); break; }
    }
    #undef LAUNCH_GEMM
  };

  auto conv = [&](const GenP& p, const float* X, float* Y, int kout, int om) {
    fill0(mbuf, 3 * N * H_DIM);             // m, den, agg contiguous
    fill0(stats, 2 * K2_DIM);               // sum, sumsq
    long long tot = E * (H_DIM / 4);
    int eb = nblk(tot);
    edge_pass_kernel<0><<<eb, 256, 0, stream>>>(X, ei, p.t, mbuf, den, agg, E);
    edge_pass_kernel<1><<<eb, 256, 0, stream>>>(X, ei, p.t, mbuf, den, agg, E);
    edge_pass_kernel<2><<<eb, 256, 0, stream>>>(X, ei, p.t, mbuf, den, agg, E);
    // hmid = (X + aggr) @ w1
    pack_a(1, X, agg, (int)N, H_DIM, nullptr, nullptr, nullptr, nullptr);
    pack_b(p.w1, H_DIM, K2_DIM);
    gemm(0, hmid, (int)N, H_DIM, K2_DIM, nullptr);
    bn_partial_kernel<<<512, K2_DIM, 0, stream>>>(hmid, stats, stats + K2_DIM,
                                                  (int)N, K2_DIM);
    bn_finalize_kernel<<<1, K2_DIM, 0, stream>>>(stats, stats + K2_DIM,
                                                 stats + 2 * K2_DIM,
                                                 stats + 3 * K2_DIM,
                                                 1.0f / (float)N, K2_DIM);
    // Y = act( relu(BN(hmid)) @ w2 )
    pack_a(2, hmid, nullptr, (int)N, K2_DIM,
           stats + 2 * K2_DIM, stats + 3 * K2_DIM, p.g, p.b);
    pack_b(p.w2, K2_DIM, kout);
    gemm(om, Y, (int)N, K2_DIM, kout, nullptr);
  };

  // h = x @ n_lin_w + n_lin_b
  pack_a(0, x, nullptr, (int)N, C_IN, nullptr, nullptr, nullptr, nullptr);
  pack_b(n_lin_w, C_IN, H_DIM);
  gemm(3, hbuf, (int)N, C_IN, H_DIM, n_lin_b);
  // encoder
  conv(enc0,  hbuf, hbuf, H_DIM, /*relu*/1);
  conv(enc1,  hbuf, hbuf, H_DIM, /*relu*/1);
  conv(encmu, hbuf, muo,  H_DIM, /*none*/0);
  conv(encls, hbuf, lso,  H_DIM, /*min10*/2);
  // reparametrize: Z -> second output slice
  z_kernel<<<nblk(N * H_DIM), 256, 0, stream>>>(muo, lso, noise, out_z, N * H_DIM);
  // decoder (reads Z from d_out slice)
  conv(dec0, out_z, hbuf, H_DIM, /*relu*/1);
  conv(dec1, hbuf, out_h, C_IN, /*relu*/1);
}
